// SimplifiedTransformerBlock_83090437308510
// MI455X (gfx1250) — compile-verified, matrix operand order checked by reading, not run on hardware
//
#include <hip/hip_runtime.h>
#include <math.h>

// ---------------- CDNA5 bf16 WMMA plumbing ----------------
typedef __bf16 bf16;
typedef __attribute__((ext_vector_type(16))) __bf16 v16bf;
typedef __attribute__((ext_vector_type(8)))  __bf16 v8bf;
typedef __attribute__((ext_vector_type(8)))  float  v8f;

__device__ __forceinline__ v8f wmma_bf16(v16bf a, v16bf b, v8f c) {
  // D(16x16 f32) = A(16x32 bf16) * B(32x16 bf16) + C
  return __builtin_amdgcn_wmma_f32_16x16x32_bf16(
      /*neg_a=*/false, a, /*neg_b=*/false, b,
      /*c_mod=*/(short)0, c, /*reuse_a=*/false, /*reuse_b=*/false);
}

// Load a 16x32 bf16 A/B fragment. Per CDNA5 ISA layout:
//  lanes 0-15 : row = lane,    K = {0..7, 16..23}
//  lanes 16-31: row = lane-16, K = {8..15, 24..31}
// base points at element [row0][k0]; stride = row pitch in elements.
__device__ __forceinline__ v16bf load_frag(const bf16* base, int stride) {
  const int lane = threadIdx.x & 31;
  const int r  = lane & 15;
  const int kb = (lane & 16) ? 8 : 0;
  const bf16* p = base + (size_t)r * stride + kb;
  v16bf f;
#pragma unroll
  for (int i = 0; i < 8; ++i) { f[i] = p[i]; f[i + 8] = p[i + 16]; }
  return f;
}

// ---------------- Problem constants ----------------
constexpr int NB  = 2;            // batch
constexpr int LL  = 2048;         // sequence
constexpr int DD  = 1024;         // d_model
constexpr int HH  = 16;           // heads
constexpr int DK  = 64;           // head dim
constexpr int MM  = NB * LL;      // 4096 rows
constexpr int QKS = 2 * DD;       // qk row stride (2048)

// ---------------- fp32 -> bf16 convert ----------------
__global__ void f2bf_kernel(const float* __restrict__ in, bf16* __restrict__ out, int n) {
  int i = blockIdx.x * blockDim.x + threadIdx.x;
  if (i < n) out[i] = (bf16)in[i];
}

// ---------------- x transpose: xt[n][d][l] = bf16(x[n][l][d]) ----------------
// Gives the flash-attention V operand a contiguous (b128-loadable) layout.
__global__ __launch_bounds__(256) void transpose_kernel(const float* __restrict__ x,
                                                        bf16* __restrict__ xt) {
  __shared__ float tile[32][33];
  const int n  = blockIdx.z;
  const int l0 = blockIdx.x * 32;
  const int d0 = blockIdx.y * 32;
  const int tx = threadIdx.x;        // 0..31
  const int ty = threadIdx.y;        // 0..7
#pragma unroll
  for (int i = 0; i < 32; i += 8)
    tile[ty + i][tx] = x[(size_t)(n * LL + l0 + ty + i) * DD + d0 + tx];
  __syncthreads();
#pragma unroll
  for (int i = 0; i < 32; i += 8)
    xt[(size_t)(n * DD + d0 + ty + i) * LL + l0 + tx] = (bf16)tile[tx][ty + i];
}

// ---------------- column sum over L of x : xsum[n][d] ----------------
__global__ void colsum_kernel(const float* __restrict__ x, float* __restrict__ xsum) {
  int d = blockIdx.x * blockDim.x + threadIdx.x;   // 0..DD-1
  int n = blockIdx.y;
  float s = 0.f;
  for (int l = 0; l < LL; ++l) s += x[(size_t)(n * LL + l) * DD + d];
  xsum[n * DD + d] = s;
}

// ---------------- Tiled WMMA GEMM:  C = A(MxK) * B(NnxK)^T + bias ----------------
// Block tile 128x64, 256 threads = 8 waves (4x2), wave tile 32x32, K-slab 64 (8 WMMA/stage).
// EPI 0: bf16 out; EPI 1: tanh-GELU -> bf16; EPI 2: fp32 out = aff*resid + bff*(acc+bias)
template <int EPI>
__global__ __launch_bounds__(256) void gemm_wmma(
    const bf16* __restrict__ A, const bf16* __restrict__ B,
    const float* __restrict__ bias,
    bf16* __restrict__ outb, float* __restrict__ outf,
    const float* __restrict__ resid,
    const float* __restrict__ aff_p, const float* __restrict__ bff_p,
    int M, int Nn, int K) {
  __shared__ __align__(16) bf16 As[128 * 72];   // 128x64, padded row stride 72 (144B)
  __shared__ __align__(16) bf16 Bs[64 * 72];    // 64x64

  const int tid  = threadIdx.x;
  const int lane = tid & 31;
  const int wid  = tid >> 5;
  const int wm   = wid & 3;          // wave row  (0..3)
  const int wn   = wid >> 2;         // wave col  (0..1)
  const int rowBase = blockIdx.y * 128;
  const int colBase = blockIdx.x * 64;

  const v8f z = {0.f,0.f,0.f,0.f,0.f,0.f,0.f,0.f};
  v8f acc[2][2] = {{z, z}, {z, z}};

  const int ar = tid >> 1;           // A stage: 128 rows x 2 half-rows of 32 bf16
  const int ac = (tid & 1) * 32;
  const int br = tid >> 2;           // B stage: 64 rows x 4 chunks of 16 bf16
  const int bc = (tid & 3) * 16;

  for (int k0 = 0; k0 < K; k0 += 64) {
    const bf16* ag = A + (size_t)(rowBase + ar) * K + k0 + ac;
    *(v8bf*)&As[ar * 72 + ac]      = *(const v8bf*)(ag);
    *(v8bf*)&As[ar * 72 + ac + 8]  = *(const v8bf*)(ag + 8);
    *(v8bf*)&As[ar * 72 + ac + 16] = *(const v8bf*)(ag + 16);
    *(v8bf*)&As[ar * 72 + ac + 24] = *(const v8bf*)(ag + 24);
    const bf16* bg = B + (size_t)(colBase + br) * K + k0 + bc;
    *(v8bf*)&Bs[br * 72 + bc]     = *(const v8bf*)(bg);
    *(v8bf*)&Bs[br * 72 + bc + 8] = *(const v8bf*)(bg + 8);
    if (k0 + 64 < K) {               // hint next K slab into cache (global_prefetch_b8)
      __builtin_prefetch(ag + 64, 0, 1);
      __builtin_prefetch(bg + 64, 0, 1);
    }
    __syncthreads();

#pragma unroll
    for (int s = 0; s < 2; ++s) {
      v16bf af0 = load_frag(&As[(wm * 32 +  0) * 72 + s * 32], 72);
      v16bf af1 = load_frag(&As[(wm * 32 + 16) * 72 + s * 32], 72);
      v16bf bf0 = load_frag(&Bs[(wn * 32 +  0) * 72 + s * 32], 72);
      v16bf bf1 = load_frag(&Bs[(wn * 32 + 16) * 72 + s * 32], 72);
      acc[0][0] = wmma_bf16(af0, bf0, acc[0][0]);
      acc[0][1] = wmma_bf16(af0, bf1, acc[0][1]);
      acc[1][0] = wmma_bf16(af1, bf0, acc[1][0]);
      acc[1][1] = wmma_bf16(af1, bf1, acc[1][1]);
    }
    __syncthreads();
  }

  // C/D layout: reg r -> row r (+8 for lanes 16-31), col = lane&15
  const int hi8 = (lane & 16) ? 8 : 0;
  const int cl  = lane & 15;
  float aff = 0.f, bff = 0.f;
  if (EPI == 2) { aff = *aff_p; bff = *bff_p; }

#pragma unroll
  for (int i = 0; i < 2; ++i) {
#pragma unroll
    for (int j = 0; j < 2; ++j) {
      const int gcol = colBase + wn * 32 + j * 16 + cl;
      const float bv = bias[gcol];
#pragma unroll
      for (int r = 0; r < 8; ++r) {
        const int grow = rowBase + wm * 32 + i * 16 + hi8 + r;
        const float v = acc[i][j][r] + bv;
        const size_t idx = (size_t)grow * Nn + gcol;
        if (EPI == 0) {
          outb[idx] = (bf16)v;
        } else if (EPI == 1) {
          const float g = 0.5f * v * (1.f + tanhf(0.7978845608f * (v + 0.044715f * v * v * v)));
          outb[idx] = (bf16)g;
        } else {
          outf[idx] = aff * resid[idx] + bff * v;
        }
      }
    }
  }
}

// ---------------- Flash attention (one wave handles 16 query rows of one head) ----------------
// out = beta*softmax(Q K^T/8)@V - (gamma/L)*colsum(V) + alpha*V_row   (V = x reshaped per head)
__global__ __launch_bounds__(32) void attn_flash(
    const bf16* __restrict__ qk, const bf16* __restrict__ xt,
    const float* __restrict__ x, const float* __restrict__ xsum,
    float* __restrict__ attn,
    const float* __restrict__ alpha_p, const float* __restrict__ beta_p,
    const float* __restrict__ gamma_p) {
  const int lane = threadIdx.x & 31;
  const int n  = blockIdx.y >> 4;
  const int h  = blockIdx.y & 15;
  const int l0 = blockIdx.x * 16;
  const int hi8 = (lane & 16) ? 8 : 0;
  const int cl  = lane & 15;
  __shared__ __align__(16) bf16 P[16 * 32];   // exp(scores) repack buffer (accum -> A layout)

  // Q fragment is loop-invariant: 16 rows x 64 dk = two 16x32 A fragments
  v16bf qf[2];
#pragma unroll
  for (int s = 0; s < 2; ++s)
    qf[s] = load_frag(qk + (size_t)(n * LL + l0) * QKS + h * DK + s * 32, QKS);

  const v8f z = {0.f,0.f,0.f,0.f,0.f,0.f,0.f,0.f};
  v8f oacc[4] = {z, z, z, z};                 // 16 rows x 64 dk accumulator
  float m_run[8], l_run[8];
#pragma unroll
  for (int j = 0; j < 8; ++j) { m_run[j] = -1e30f; l_run[j] = 0.f; }

  for (int m0 = 0; m0 < LL; m0 += 32) {
    // scores: two 16x16 tiles (cols m0..m0+15, m0+16..m0+31), K accumulated over dk=64
    v8f s0 = z, s1 = z;
#pragma unroll
    for (int s = 0; s < 2; ++s) {
      v16bf kf0 = load_frag(qk + (size_t)(n * LL + m0) * QKS + DD + h * DK + s * 32, QKS);
      s0 = wmma_bf16(qf[s], kf0, s0);
      v16bf kf1 = load_frag(qk + (size_t)(n * LL + m0 + 16) * QKS + DD + h * DK + s * 32, QKS);
      s1 = wmma_bf16(qf[s], kf1, s1);
    }

    // online softmax per row (rows striped: reg j = row j + hi8; cols across 16-lane halves)
#pragma unroll
    for (int j = 0; j < 8; ++j) {
      float a = s0[j] * 0.125f;     // 1/sqrt(dk)
      float b = s1[j] * 0.125f;
      float mx = fmaxf(a, b);
#pragma unroll
      for (int off = 1; off < 16; off <<= 1)
        mx = fmaxf(mx, __shfl_xor(mx, off, 32));
      const float mnew = fmaxf(m_run[j], mx);
      const float corr = __expf(m_run[j] - mnew);
      const float p0 = __expf(a - mnew);
      const float p1 = __expf(b - mnew);
      float rs = p0 + p1;
#pragma unroll
      for (int off = 1; off < 16; off <<= 1)
        rs += __shfl_xor(rs, off, 32);
      l_run[j] = l_run[j] * corr + rs;
      m_run[j] = mnew;
#pragma unroll
      for (int t = 0; t < 4; ++t) oacc[t][j] *= corr;
      P[(j + hi8) * 32 + cl]      = (bf16)p0;
      P[(j + hi8) * 32 + cl + 16] = (bf16)p1;
    }
    __syncthreads();

    // P(16x32) @ V(32x64): V from transposed xt[n][d][l] -> contiguous b128 fragment loads
    v16bf pf = load_frag(P, 32);
#pragma unroll
    for (int t = 0; t < 4; ++t) {
      v16bf vf = load_frag(xt + (size_t)(n * DD + h * DK + t * 16) * LL + m0, LL);
      oacc[t] = wmma_bf16(pf, vf, oacc[t]);
    }
    __syncthreads();
  }

  const float alpha = *alpha_p, beta = *beta_p, gamma = *gamma_p;
  const float gl = gamma * (1.f / (float)LL);
#pragma unroll
  for (int t = 0; t < 4; ++t) {
#pragma unroll
    for (int j = 0; j < 8; ++j) {
      const int row = l0 + j + hi8;
      const int d   = h * DK + t * 16 + cl;
      const size_t idx = (size_t)(n * LL + row) * DD + d;
      attn[idx] = beta * (oacc[t][j] / l_run[j]) - gl * xsum[n * DD + d] + alpha * x[idx];
    }
  }
}

// ---------------- host-side orchestration ----------------
extern "C" void kernel_launch(void* const* d_in, const int* in_sizes, int n_in,
                              void* d_out, int out_size, void* d_ws, size_t ws_size,
                              hipStream_t stream) {
  const float* x     = (const float*)d_in[0];   // (2,2048,1024)
  const float* Wqk   = (const float*)d_in[1];   // (2048,1024)
  const float* bqk   = (const float*)d_in[2];   // (2048,)
  const float* W1    = (const float*)d_in[3];   // (4096,1024)
  const float* b1    = (const float*)d_in[4];   // (4096,)
  const float* W2    = (const float*)d_in[5];   // (1024,4096)
  const float* b2    = (const float*)d_in[6];   // (1024,)
  const float* alpha = (const float*)d_in[7];
  const float* beta  = (const float*)d_in[8];
  const float* gamma = (const float*)d_in[9];
  const float* a_ff  = (const float*)d_in[10];
  const float* b_ff  = (const float*)d_in[11];
  float* out = (float*)d_out;
  (void)in_sizes; (void)n_in; (void)out_size; (void)ws_size;

  // workspace carve-out (~105 MB, 256B-aligned chunks)
  char* ws = (char*)d_ws;
  size_t off = 0;
  auto carve = [&](size_t bytes) -> char* {
    char* p = ws + off;
    off += (bytes + 255) & ~(size_t)255;
    return p;
  };
  bf16*  xbf   = (bf16*)carve((size_t)MM * DD * 2);          // 8 MB  (x, row-major bf16)
  bf16*  xt    = (bf16*)carve((size_t)MM * DD * 2);          // 8 MB  (x, [n][d][l] bf16)
  bf16*  wqkbf = (bf16*)carve((size_t)QKS * DD * 2);         // 4 MB
  bf16*  w1bf  = (bf16*)carve((size_t)4 * DD * DD * 2);      // 8 MB
  bf16*  w2bf  = (bf16*)carve((size_t)4 * DD * DD * 2);      // 8 MB
  bf16*  qkbf  = (bf16*)carve((size_t)MM * QKS * 2);         // 16 MB
  bf16*  h1bf  = (bf16*)carve((size_t)MM * 4 * DD * 2);      // 32 MB
  float* attnb = (float*)carve((size_t)MM * DD * 4);         // 16 MB
  float* xsum  = (float*)carve((size_t)NB * DD * 4);         // 8 KB

  const int CT = 256;
  auto cvt = [&](const float* src, bf16* dst, size_t n) {
    f2bf_kernel<<<dim3((unsigned)((n + CT - 1) / CT)), dim3(CT), 0, stream>>>(src, dst, (int)n);
  };
  cvt(x,   xbf,   (size_t)MM * DD);
  cvt(Wqk, wqkbf, (size_t)QKS * DD);
  cvt(W1,  w1bf,  (size_t)4 * DD * DD);
  cvt(W2,  w2bf,  (size_t)4 * DD * DD);

  transpose_kernel<<<dim3(LL / 32, DD / 32, NB), dim3(32, 8), 0, stream>>>(x, xt);
  colsum_kernel<<<dim3(DD / CT, NB), dim3(CT), 0, stream>>>(x, xsum);

  // 1) qk = x @ Wqk^T + bqk   (4096 x 2048, K=1024) -> bf16
  gemm_wmma<0><<<dim3(QKS / 64, MM / 128), dim3(256), 0, stream>>>(
      xbf, wqkbf, bqk, qkbf, nullptr, nullptr, nullptr, nullptr, MM, QKS, DD);

  // 2) flash attention + distributed (beta, -gamma/L, alpha*I) epilogue -> fp32
  attn_flash<<<dim3(LL / 16, NB * HH), dim3(32), 0, stream>>>(
      qkbf, xt, x, xsum, attnb, alpha, beta, gamma);

  // 3) h1 = gelu(x @ W1^T + b1)   (4096 x 4096, K=1024) -> bf16
  gemm_wmma<1><<<dim3(4 * DD / 64, MM / 128), dim3(256), 0, stream>>>(
      xbf, w1bf, b1, h1bf, nullptr, nullptr, nullptr, nullptr, MM, 4 * DD, DD);

  // 4) out = alpha_ff*attn + beta_ff*(h1 @ W2^T + b2)   (4096 x 1024, K=4096) -> fp32
  gemm_wmma<2><<<dim3(DD / 64, MM / 128), dim3(256), 0, stream>>>(
      h1bf, w2bf, b2, nullptr, out, attnb, a_ff, b_ff, MM, DD, 4 * DD);
}
